// SelfAttentionHead_10204842295599
// MI455X (gfx1250) — matrix operand order repaired and checked by reading, not compiled, not run on hardware
//
#include <hip/hip_runtime.h>

typedef __attribute__((ext_vector_type(16))) __bf16 v16bf;
typedef __attribute__((ext_vector_type(8)))  float  v8f;

#define B_ 8
#define T_ 2048
#define C_ 1024
#define H_ 64
#define QSCALE 0.03125f                    // C^-0.5 = 1/32, folded into q
#define L2E 1.44269504088896340736f

union BF16x16 { v16bf v; uint4 q[2]; };
union BF16x8  { uint4 q; __bf16 e[8]; };

static __device__ __forceinline__ v8f wmma_bf16(v16bf a, v16bf b, v8f c) {
  // (neg_a, A, neg_b, B, c_mod, C, reuse_a, reuse_b)
  return __builtin_amdgcn_wmma_f32_16x16x32_bf16(false, a, false, b, (short)0, c,
                                                 false, false);
}

static __device__ __forceinline__ v16bf load_frag(const __bf16* p0, const __bf16* p1) {
  BF16x16 u;
  u.q[0] = *(const uint4*)p0;
  u.q[1] = *(const uint4*)p1;
  return u.v;
}

static __device__ __forceinline__ float rmax16(float v) {
  #pragma unroll
  for (int m = 1; m < 16; m <<= 1) v = fmaxf(v, __shfl_xor(v, m, 32));
  return v;
}

static __device__ __forceinline__ v16bf ones_frag() {
  BF16x16 u;
  #pragma unroll
  for (int i = 0; i < 16; ++i) u.v[i] = (__bf16)1.0f;
  return u.v;
}

// ---------------------------------------------------------------------------
// Kernel 1: W[C,H] f32  ->  WT[H,C] bf16  (so WMMA B-fragments load contiguous)
// ---------------------------------------------------------------------------
__global__ void wt_transpose_kernel(const float* __restrict__ Wq,
                                    const float* __restrict__ Wk,
                                    const float* __restrict__ Wv,
                                    __bf16* __restrict__ WTq,
                                    __bf16* __restrict__ WTk,
                                    __bf16* __restrict__ WTv) {
  int idx = blockIdx.x * blockDim.x + threadIdx.x;   // 3*C*H threads
  int w   = idx / (C_ * H_);
  int r   = idx % (C_ * H_);
  int c   = r / H_;
  int h   = r % H_;
  const float* src = (w == 0) ? Wq : (w == 1) ? Wk : Wv;
  __bf16*      dst = (w == 0) ? WTq : (w == 1) ? WTk : WTv;
  dst[h * C_ + c] = (__bf16)src[c * H_ + h];
}

// ---------------------------------------------------------------------------
// Kernel 2: fused QKV projection. One wave = 16 rows x all 192 output cols.
// Stores q (scaled) / k as bf16 [B*T, H]; v transposed as bf16 [B, H, T].
// ---------------------------------------------------------------------------
__global__ void qkv_proj_kernel(const float*  __restrict__ x,
                                const __bf16* __restrict__ WTq,
                                const __bf16* __restrict__ WTk,
                                const __bf16* __restrict__ WTv,
                                __bf16* __restrict__ qb,
                                __bf16* __restrict__ kb,
                                __bf16* __restrict__ vT) {
  const int lane = threadIdx.x & 31;
  const int wave = threadIdx.x >> 5;
  const int hi   = (lane >> 4) & 1;
  const int ln   = lane & 15;
  const int row0 = blockIdx.x * 128 + wave * 16;     // global row (B*T flat)

  v8f aq[4], ak[4], av[4];
  #pragma unroll
  for (int nt = 0; nt < 4; ++nt) {
    #pragma unroll
    for (int r = 0; r < 8; ++r) { aq[nt][r] = 0.f; ak[nt][r] = 0.f; av[nt][r] = 0.f; }
  }

  const float* xr = x + (size_t)(row0 + ln) * C_;

  for (int kt = 0; kt < C_ / 32; ++kt) {
    // A fragment: 16x32 bf16 tile of x at K-window kt*32
    const int kbA = kt * 32 + hi * 16;
    float4 f0 = *(const float4*)(xr + kbA);
    float4 f1 = *(const float4*)(xr + kbA + 4);
    float4 f2 = *(const float4*)(xr + kbA + 8);
    float4 f3 = *(const float4*)(xr + kbA + 12);
    BF16x16 a;
    a.v[0] = (__bf16)f0.x; a.v[1] = (__bf16)f0.y; a.v[2]  = (__bf16)f0.z; a.v[3]  = (__bf16)f0.w;
    a.v[4] = (__bf16)f1.x; a.v[5] = (__bf16)f1.y; a.v[6]  = (__bf16)f1.z; a.v[7]  = (__bf16)f1.w;
    a.v[8] = (__bf16)f2.x; a.v[9] = (__bf16)f2.y; a.v[10] = (__bf16)f2.z; a.v[11] = (__bf16)f2.w;
    a.v[12] = (__bf16)f3.x; a.v[13] = (__bf16)f3.y; a.v[14] = (__bf16)f3.z; a.v[15] = (__bf16)f3.w;

    const int kbB = kt * 32 + hi * 16;               // B frag: 16 contiguous K
    // group the 4 B-fragments per weight so loads clause + overlap the WMMAs
    v16bf bq[4], bk2[4], bv[4];
    #pragma unroll
    for (int nt = 0; nt < 4; ++nt) {
      const __bf16* p = WTq + (size_t)(nt * 16 + ln) * C_ + kbB;
      bq[nt] = load_frag(p, p + 8);
    }
    #pragma unroll
    for (int nt = 0; nt < 4; ++nt) aq[nt] = wmma_bf16(a.v, bq[nt], aq[nt]);
    #pragma unroll
    for (int nt = 0; nt < 4; ++nt) {
      const __bf16* p = WTk + (size_t)(nt * 16 + ln) * C_ + kbB;
      bk2[nt] = load_frag(p, p + 8);
    }
    #pragma unroll
    for (int nt = 0; nt < 4; ++nt) ak[nt] = wmma_bf16(a.v, bk2[nt], ak[nt]);
    #pragma unroll
    for (int nt = 0; nt < 4; ++nt) {
      const __bf16* p = WTv + (size_t)(nt * 16 + ln) * C_ + kbB;
      bv[nt] = load_frag(p, p + 8);
    }
    #pragma unroll
    for (int nt = 0; nt < 4; ++nt) av[nt] = wmma_bf16(a.v, bv[nt], av[nt]);
  }

  // store q (scaled) & k row-major, v transposed [B,H,T]
  const int gb = row0 >> 11;                // batch (T_ = 2048)
  const int tb = (row0 & (T_ - 1)) + hi * 8;
  #pragma unroll
  for (int nt = 0; nt < 4; ++nt) {
    const int h = nt * 16 + ln;
    #pragma unroll
    for (int r = 0; r < 8; ++r) {
      size_t ro = (size_t)(row0 + hi * 8 + r) * H_ + h;
      qb[ro] = (__bf16)(aq[nt][r] * QSCALE);
      kb[ro] = (__bf16)(ak[nt][r]);
    }
    BF16x8 pk;
    #pragma unroll
    for (int r = 0; r < 8; ++r) pk.e[r] = (__bf16)(av[nt][r]);
    *(uint4*)(vT + (size_t)gb * H_ * T_ + (size_t)h * T_ + tb) = pk.q;
  }
}

// ---------------------------------------------------------------------------
// Kernel 3: flash attention, 64-key blocks, online softmax.
// Row-sum is done on the matrix pipe (P @ ones); only row-max uses shuffles.
// ---------------------------------------------------------------------------
__global__ void attn_kernel(const __bf16* __restrict__ qb,
                            const __bf16* __restrict__ kb,
                            const __bf16* __restrict__ vT,
                            float* __restrict__ out) {
  __shared__ __bf16 pl[4 * 16 * 64];                 // per-wave P staging (8 KB)
  const int lane = threadIdx.x & 31;
  const int wave = threadIdx.x >> 5;
  const int hi   = (lane >> 4) & 1;
  const int ln   = lane & 15;
  const int b    = blockIdx.x >> 5;
  const int m0   = (blockIdx.x & 31) * 64 + wave * 16;   // query row base in batch

  // Q A-fragments: K(=H) windows 0..31 and 32..63
  const __bf16* qr = qb + (size_t)(b * T_ + m0 + ln) * H_;
  const v16bf a0 = load_frag(qr + hi * 16,      qr + 8 + hi * 16);
  const v16bf a1 = load_frag(qr + 32 + hi * 16, qr + 40 + hi * 16);
  const v16bf ones = ones_frag();

  v8f o[4];
  float mrow[8], lrow[8];
  #pragma unroll
  for (int nt = 0; nt < 4; ++nt)
    #pragma unroll
    for (int r = 0; r < 8; ++r) o[nt][r] = 0.f;
  #pragma unroll
  for (int r = 0; r < 8; ++r) { mrow[r] = -1e30f; lrow[r] = 0.f; }

  __bf16* pw = pl + wave * 1024;
  const __bf16* vbase = vT + (size_t)b * H_ * T_;
  const __bf16* kbb   = kb + (size_t)b * T_ * H_;

  for (int s0 = 0; s0 <= m0 + 15; s0 += 64) {        // 64-key blocks
    // prefetch next block's K rows and V columns (global_prefetch_b8)
    if (s0 + 64 <= m0 + 15) {
      const int s1 = s0 + 64;
      __builtin_prefetch(kbb + (size_t)(s1 + ln) * H_, 0, 3);
      __builtin_prefetch(kbb + (size_t)(s1 + 16 + ln) * H_, 0, 3);
      __builtin_prefetch(kbb + (size_t)(s1 + 32 + ln) * H_, 0, 3);
      __builtin_prefetch(kbb + (size_t)(s1 + 48 + ln) * H_, 0, 3);
      __builtin_prefetch(vbase + (size_t)ln * T_ + s1, 0, 3);
      __builtin_prefetch(vbase + (size_t)(16 + ln) * T_ + s1, 0, 3);
      __builtin_prefetch(vbase + (size_t)(32 + ln) * T_ + s1, 0, 3);
      __builtin_prefetch(vbase + (size_t)(48 + ln) * T_ + s1, 0, 3);
    }

    // S = Q K^T for 4 key n-tiles (64 keys)
    v8f st[4];
    #pragma unroll
    for (int nt = 0; nt < 4; ++nt) {
      const __bf16* kr = kbb + (size_t)(s0 + nt * 16 + ln) * H_;
      v16bf blo = load_frag(kr + hi * 16,      kr + hi * 16 + 8);   // h 0..31
      v16bf bhi = load_frag(kr + 32 + hi * 16, kr + 40 + hi * 16);  // h 32..63
      v8f z;
      #pragma unroll
      for (int r = 0; r < 8; ++r) z[r] = 0.f;
      st[nt] = wmma_bf16(a0, blo, wmma_bf16(a1, bhi, z));
    }

    if (s0 + 63 > m0) {                              // causal mask near diagonal
      #pragma unroll
      for (int nt = 0; nt < 4; ++nt)
        #pragma unroll
        for (int r = 0; r < 8; ++r)
          if (s0 + nt * 16 + ln > m0 + hi * 8 + r) st[nt][r] = -1e30f;
    }

    // online softmax: row max (shuffle tree), P = exp2((S-m)*log2e)
    float sc[8];
    #pragma unroll
    for (int r = 0; r < 8; ++r) {
      float mx = fmaxf(fmaxf(st[0][r], st[1][r]), fmaxf(st[2][r], st[3][r]));
      mx = rmax16(mx);
      float mn = fmaxf(mrow[r], mx);
      sc[r] = exp2f((mrow[r] - mn) * L2E);
      mrow[r] = mn;
      const int m = (r + hi * 8) * 64 + ln;
      #pragma unroll
      for (int nt = 0; nt < 4; ++nt) {
        float p = exp2f((st[nt][r] - mn) * L2E);
        pw[m + nt * 16] = (__bf16)p;                 // restripe via LDS
      }
      o[0][r] *= sc[r]; o[1][r] *= sc[r]; o[2][r] *= sc[r]; o[3][r] *= sc[r];
    }
    asm volatile("s_wait_dscnt 0" ::: "memory");

    // reload P as A-fragments (keys 0..31 / 32..63 of this block)
    const __bf16* pr = pw + ln * 64 + hi * 16;
    v16bf pa0 = load_frag(pr,      pr + 8);
    v16bf pa1 = load_frag(pr + 32, pr + 40);

    // row sums on the matrix pipe: l_blk = P @ ones (replicated per row)
    v8f z2;
    #pragma unroll
    for (int r = 0; r < 8; ++r) z2[r] = 0.f;
    v8f ls = wmma_bf16(pa0, ones, wmma_bf16(pa1, ones, z2));
    #pragma unroll
    for (int r = 0; r < 8; ++r) lrow[r] = lrow[r] * sc[r] + ls[r];

    // O += P * V ; vT is contiguous along t (= K dim of PV)
    #pragma unroll
    for (int nt = 0; nt < 4; ++nt) {
      const __bf16* vp = vbase + (size_t)(nt * 16 + ln) * T_ + s0 + hi * 16;
      v16bf v0 = load_frag(vp,      vp + 8);         // keys s0..s0+31
      v16bf v1 = load_frag(vp + 32, vp + 40);        // keys s0+32..s0+63
      o[nt] = wmma_bf16(pa0, v0, wmma_bf16(pa1, v1, o[nt]));
    }
  }

  #pragma unroll
  for (int r = 0; r < 8; ++r) {
    const float inv = 1.0f / lrow[r];
    const size_t ro = (size_t)(b * T_ + m0 + hi * 8 + r) * H_;
    #pragma unroll
    for (int nt = 0; nt < 4; ++nt)
      out[ro + nt * 16 + ln] = o[nt][r] * inv;
  }
}

// ---------------------------------------------------------------------------
extern "C" void kernel_launch(void* const* d_in, const int* in_sizes, int n_in,
                              void* d_out, int out_size, void* d_ws, size_t ws_size,
                              hipStream_t stream) {
  (void)in_sizes; (void)n_in; (void)out_size; (void)ws_size;
  const float* x  = (const float*)d_in[0];
  const float* Wk = (const float*)d_in[1];
  const float* Wq = (const float*)d_in[2];
  const float* Wv = (const float*)d_in[3];
  float* out = (float*)d_out;

  __bf16* qb  = (__bf16*)d_ws;                       // [B*T, H] bf16, pre-scaled
  __bf16* kb  = qb + (size_t)B_ * T_ * H_;           // [B*T, H] bf16
  __bf16* vT  = kb + (size_t)B_ * T_ * H_;           // [B, H, T] bf16
  __bf16* WTq = vT + (size_t)B_ * T_ * H_;           // [H, C] bf16
  __bf16* WTk = WTq + (size_t)C_ * H_;
  __bf16* WTv = WTk + (size_t)C_ * H_;

  wt_transpose_kernel<<<(3 * C_ * H_) / 256, 256, 0, stream>>>(Wq, Wk, Wv, WTq, WTk, WTv);
  qkv_proj_kernel<<<(B_ * T_) / 128, 256, 0, stream>>>(x, WTq, WTk, WTv, qb, kb, vT);
  attn_kernel<<<B_ * (T_ / 64), 128, 0, stream>>>(qb, kb, vT, out);
}
